// StyleConv3d_44418551776061
// MI455X (gfx1250) — compile-verified, matrix operand order checked by reading, not run on hardware
//
#include <hip/hip_runtime.h>

// ---------------------------------------------------------------------------
// StyleConv3d for MI455X (gfx1250): implicit-GEMM conv via bf16 WMMA.
//   M = t (16), N = cout tile (16), K = cin chunk (32), looped over 9 taps.
//   Weights staged to LDS with gfx1250 async copies (ASYNCcnt path).
// ---------------------------------------------------------------------------

typedef __attribute__((ext_vector_type(16))) __bf16 v16bf;
typedef __attribute__((ext_vector_type(8)))  __bf16 v8bf;
typedef __attribute__((ext_vector_type(8)))  float  v8f;
typedef __attribute__((ext_vector_type(4)))  float  v4f;

#define NB    4
#define CIN   128
#define COUT  128
#define HDIM  64
#define WDIM  64
#define TDIM  16
#define SSIZE 512
#define WTILE 8
#define PATCH_C (WTILE + 2)                    // 10 input cols (halo)
#define LDSX_HALF (3 * PATCH_C * TDIM * CIN)   // 61440 ushorts = 120 KB
#define LDSW_HALF (COUT * CIN)                 // 16384 ushorts =  32 KB

static __device__ __forceinline__ unsigned short f2bf(float f) {
  unsigned int u = __float_as_uint(f);
  u += 0x7fffu + ((u >> 16) & 1u);             // round-to-nearest-even
  return (unsigned short)(u >> 16);
}

// ---- s[b,i] = style[b,:] . style_w[i,:] + style_b[i] ----------------------
__global__ void k_style(const float* __restrict__ style,
                        const float* __restrict__ sw,
                        const float* __restrict__ sb,
                        float* __restrict__ s) {
  int tid = blockIdx.x * blockDim.x + threadIdx.x;   // 512 total
  int b = tid >> 7, i = tid & 127;
  const float* st = style + b * SSIZE;
  const float* w  = sw + i * SSIZE;
  float acc = 0.f;
  for (int j = 0; j < SSIZE; j += 4)
    acc += st[j] * w[j] + st[j+1] * w[j+1] + st[j+2] * w[j+2] + st[j+3] * w[j+3];
  s[tid] = acc + sb[i];
}

// ---- wsq[o,i] = sum_k W[o,i,k]^2 ; wb[tap][o][i] = bf16(W) ----------------
__global__ void k_prep_w(const float* __restrict__ wt,
                         float* __restrict__ wsq,
                         unsigned short* __restrict__ wb) {
  int tid = blockIdx.x * blockDim.x + threadIdx.x;   // 16384 total
  int o = tid >> 7, i = tid & 127;
  const float* wp = wt + (o * CIN + i) * 9;
  float acc = 0.f;
#pragma unroll
  for (int k = 0; k < 9; ++k) {
    float v = wp[k];
    acc += v * v;
    wb[k * (COUT * CIN) + o * CIN + i] = f2bf(v);
  }
  wsq[o * CIN + i] = acc;
}

// ---- sig_inv[b,o] = rsqrt(sum_i wsq[o,i]*s[b,i]^2 + 1e-8) -----------------
__global__ void k_siginv(const float* __restrict__ wsq,
                         const float* __restrict__ s,
                         float* __restrict__ siginv) {
  int tid = blockIdx.x * blockDim.x + threadIdx.x;   // 512 total
  int b = tid >> 7, o = tid & 127;
  const float* q  = wsq + o * CIN;
  const float* sv = s + b * CIN;
  float acc = 0.f;
  for (int i = 0; i < CIN; ++i) acc += q[i] * sv[i] * sv[i];
  siginv[tid] = rsqrtf(acc + 1e-8f);
}

// ---- main fused conv: one block = (b, h, 8 w-cols) x all 128 couts --------
__global__ void __launch_bounds__(256)
k_conv(const float* __restrict__ x, const float* __restrict__ noise,
       const unsigned short* __restrict__ wb, const float* __restrict__ s,
       const float* __restrict__ siginv, const float* __restrict__ np,
       const float* __restrict__ bp, float* __restrict__ out) {
  extern __shared__ unsigned short smem[];
  unsigned short* lx = smem;                // [3][10][16][128] bf16, cin innermost
  unsigned short* lw = smem + LDSX_HALF;    // [o(128)][i(128)] bf16

  int bid = blockIdx.x;
  int w0  = (bid & 7) * WTILE;
  int hh  = (bid >> 3) & 63;
  int bb  = bid >> 9;
  int tid = threadIdx.x;

  // ---- stage modulated input patch x*s -> LDS bf16 ----
  for (int idx = tid; idx < 3 * PATCH_C * CIN; idx += 256) {   // 3840/256 = 15 iters
    int r   = idx / (PATCH_C * CIN);
    int rem = idx - r * (PATCH_C * CIN);
    int c   = rem >> 7;
    int i   = rem & 127;
    int gh = hh + r - 1, gw = w0 + c - 1;
    unsigned short* dst = lx + ((r * PATCH_C + c) * TDIM) * CIN + i;
    if ((unsigned)gh < (unsigned)HDIM && (unsigned)gw < (unsigned)WDIM) {
      float sv = s[bb * CIN + i];
      const float* px = x + ((((size_t)bb * CIN + i) * HDIM + gh) * WDIM + gw) * TDIM;
      v4f a0 = *(const v4f*)(px);
      v4f a1 = *(const v4f*)(px + 4);
      v4f a2 = *(const v4f*)(px + 8);
      v4f a3 = *(const v4f*)(px + 12);
#pragma unroll
      for (int t = 0; t < 4; ++t) dst[t * CIN]        = f2bf(a0[t] * sv);
#pragma unroll
      for (int t = 0; t < 4; ++t) dst[(t + 4) * CIN]  = f2bf(a1[t] * sv);
#pragma unroll
      for (int t = 0; t < 4; ++t) dst[(t + 8) * CIN]  = f2bf(a2[t] * sv);
#pragma unroll
      for (int t = 0; t < 4; ++t) dst[(t + 12) * CIN] = f2bf(a3[t] * sv);
    } else {
#pragma unroll
      for (int t = 0; t < TDIM; ++t) dst[t * CIN] = 0;
    }
  }

  int wv   = tid >> 5;       // wave id = output column within tile
  int lane = tid & 31;
  int l    = lane & 15;
  int hi   = lane >> 4;

  v8f zero = {0.f, 0.f, 0.f, 0.f, 0.f, 0.f, 0.f, 0.f};
  v8f acc[8];
#pragma unroll
  for (int n = 0; n < 8; ++n) acc[n] = zero;

  unsigned int lw_base = (unsigned int)(uintptr_t)lw;  // flat->LDS addr truncation

  for (int tap = 0; tap < 9; ++tap) {
    __syncthreads();  // previous tap's lw reads done; lx ready on first pass
    // ---- stage this tap's weights (32 KB) via async memory->LDS copies ----
    {
      unsigned long long gbase =
          (unsigned long long)(uintptr_t)(wb + tap * (COUT * CIN));
#pragma unroll
      for (int k = 0; k < 8; ++k) {            // 2048 b128 chunks / 256 threads
        int idx = tid + k * 256;
        unsigned long long ga = gbase + (unsigned long long)idx * 16u;
        unsigned int la = lw_base + (unsigned int)idx * 16u;
        asm volatile("global_load_async_to_lds_b128 %0, %1, off"
                     :: "v"(la), "v"(ga) : "memory");
      }
      if (tap < 8)
        __builtin_prefetch(wb + (tap + 1) * (COUT * CIN) + tid * 64, 0, 0);
      asm volatile("s_wait_asynccnt 0x0" ::: "memory");
    }
    __syncthreads();

    int r = tap / 3;
    int c = wv + (tap % 3);
    const unsigned short* abase = lx + ((r * PATCH_C + c) * TDIM + l) * CIN + hi * 8;
#pragma unroll
    for (int i0 = 0; i0 < CIN; i0 += 32) {
      // A fragment: lanes 0-15 hold K{0..7,16..23}, lanes 16-31 K{8..15,24..31}
      v8bf alo = *(const v8bf*)(abase + i0);        // ds_load_b128
      v8bf ahi = *(const v8bf*)(abase + i0 + 16);   // ds_load_b128
      v16bf a = __builtin_shufflevector(alo, ahi, 0,1,2,3,4,5,6,7,8,9,10,11,12,13,14,15);
      // Preload ALL 8 B fragments first so the scheduler can overlap
      // outstanding ds_loads with WMMA issue (partial s_wait_dscnt).
      v16bf bm[8];
#pragma unroll
      for (int n = 0; n < 8; ++n) {
        const unsigned short* bbase = lw + (n * 16 + l) * CIN + i0 + hi * 8;
        v8bf blo = *(const v8bf*)(bbase);
        v8bf bhi = *(const v8bf*)(bbase + 16);
        bm[n] = __builtin_shufflevector(blo, bhi, 0,1,2,3,4,5,6,7,8,9,10,11,12,13,14,15);
      }
#pragma unroll
      for (int n = 0; n < 8; ++n) {
        acc[n] = __builtin_amdgcn_wmma_f32_16x16x32_bf16(
            /*neg_a=*/false, a, /*neg_b=*/false, bm[n],
            /*c_mod=*/(short)0, acc[n], /*reuse_a=*/false, /*reuse_b=*/false);
      }
    }
  }

  // ---- fused epilogue: y*sig_inv + noise_param*noise + bias_param ----
  int gw = w0 + wv;
  const float* nsrc = noise + (((size_t)bb * HDIM + hh) * WDIM + gw) * TDIM + hi * 8;
  v4f nz0 = *(const v4f*)(nsrc);
  v4f nz1 = *(const v4f*)(nsrc + 4);
#pragma unroll
  for (int n = 0; n < 8; ++n) {
    int o = n * 16 + l;
    float si  = siginv[bb * COUT + o];
    float npv = np[o];
    float bpv = bp[o];
    float* op = out + ((((size_t)bb * COUT + o) * HDIM + hh) * WDIM + gw) * TDIM + hi * 8;
    v4f r0, r1;
#pragma unroll
    for (int v = 0; v < 4; ++v) r0[v] = acc[n][v] * si + npv * nz0[v] + bpv;
#pragma unroll
    for (int v = 0; v < 4; ++v) r1[v] = acc[n][v + 4] * si + npv * nz1[v] + bpv;
    *(v4f*)op       = r0;   // C/D layout: VGPR v -> t = hi*8 + v, contiguous
    *(v4f*)(op + 4) = r1;
  }
}

// ---------------------------------------------------------------------------
extern "C" void kernel_launch(void* const* d_in, const int* in_sizes, int n_in,
                              void* d_out, int out_size, void* d_ws, size_t ws_size,
                              hipStream_t stream) {
  const float* x       = (const float*)d_in[0];
  const float* style   = (const float*)d_in[1];
  const float* noise   = (const float*)d_in[2];
  const float* weight  = (const float*)d_in[3];
  const float* style_w = (const float*)d_in[4];
  const float* style_b = (const float*)d_in[5];
  const float* np      = (const float*)d_in[6];
  const float* bp      = (const float*)d_in[7];
  float* out = (float*)d_out;

  // scratch layout: s[512] | siginv[512] | wsq[16384] | wb bf16 [9*128*128]
  float* ws_f   = (float*)d_ws;
  float* s_buf  = ws_f;
  float* siginv = ws_f + 512;
  float* wsq    = ws_f + 1024;
  unsigned short* wbuf = (unsigned short*)(ws_f + 1024 + COUT * CIN);

  k_style<<<2, 256, 0, stream>>>(style, style_w, style_b, s_buf);
  k_prep_w<<<64, 256, 0, stream>>>(weight, wsq, wbuf);
  k_siginv<<<2, 256, 0, stream>>>(wsq, s_buf, siginv);

  size_t smem = (size_t)(LDSX_HALF + LDSW_HALF) * sizeof(unsigned short); // ~152 KB
  int grid = NB * HDIM * (WDIM / WTILE);   // 2048 workgroups
  k_conv<<<grid, 256, smem, stream>>>(x, noise, wbuf, s_buf, siginv, np, bp, out);
}
